// Net_8718783611481
// MI455X (gfx1250) — compile-verified, matrix operand order checked by reading, not run on hardware
//
#include <hip/hip_runtime.h>
#include <hip/hip_bf16.h>
#include <math.h>

// ---------------------------------------------------------------------------
// Types for WMMA fragments (wave32, CDNA5)
// ---------------------------------------------------------------------------
typedef _Float16 v16h __attribute__((ext_vector_type(16)));
typedef _Float16 v8h  __attribute__((ext_vector_type(8)));
typedef float    v8f  __attribute__((ext_vector_type(8)));
typedef int      v4i  __attribute__((ext_vector_type(4)));

union Frag16 { v16h v; v8h h[2]; };

#define B_SZ   1024
#define T_SZ   240
#define F_SZ   180
#define H_SZ   90
#define HP     96            // H padded to 16
#define NCOL   288           // 6 jb-blocks * 3 gates * 16
#define KP_ENC 192           // F padded to 32
#define KP_DEC 96            // H padded to 32
#define MROWS  (B_SZ * T_SZ) // 245760
#define KOUT   (T_SZ * HP)   // 23040 : padded K of output projection
#define NOUTP  48            // 40 padded to 48
#define SPLITS 20            // split-K for projection (720 ksteps / 20 = 36)

#define AS1 __attribute__((address_space(1)))
#define AS3 __attribute__((address_space(3)))

// gfx1250 async global->LDS DMA path (ASYNCcnt), if the toolchain exposes it.
#if defined(__AMDGCN__) && __has_builtin(__builtin_amdgcn_global_load_async_to_lds_b128)
#define HAVE_ASYNC_LDS 1
#else
#define HAVE_ASYNC_LDS 0
#endif

#if HAVE_ASYNC_LDS
__device__ __forceinline__ void async_copy_b128(const _Float16* g, _Float16* l) {
    // Signature (from hipcc diagnostic): (int4 AS1*, int4 AS3*, imm offset, imm cpol)
    __builtin_amdgcn_global_load_async_to_lds_b128(
        (AS1 v4i*)(v4i*)(_Float16*)g,   // const_cast + bitcast + addrspacecast
        (AS3 v4i*)(v4i*)l,
        0, 0);
}
#endif

__device__ __forceinline__ void wait_async_zero() {
#if defined(__AMDGCN__)
#if __has_builtin(__builtin_amdgcn_s_wait_asynccnt)
    __builtin_amdgcn_s_wait_asynccnt(0);
#else
    asm volatile("s_wait_asynccnt 0" ::: "memory");
#endif
#endif
}

__device__ __forceinline__ float sigf(float x) { return 1.0f / (1.0f + __expf(-x)); }

// ---------------------------------------------------------------------------
// Weight packing: W_ih [4H, K] (gate order i,f,g,o; f dropped) ->
// col-major fp16 [NCOL][KP], column ncol = jb*48 + gate*16 + nn encodes
// H-column j = jb*16+nn for gate in {i,g,o}. Zero padded.
// ---------------------------------------------------------------------------
__global__ void pack_w(const float* __restrict__ W, _Float16* __restrict__ Wc,
                       int K, int KP) {
    int ncol = blockIdx.x;            // [0, 288)
    int k    = threadIdx.x;           // [0, KP)
    int jb = ncol / 48, rem = ncol % 48, gate = rem / 16, nn = rem % 16;
    int j = jb * 16 + nn;
    int grow = (gate == 0) ? j : (gate == 1) ? (2 * H_SZ + j) : (3 * H_SZ + j);
    float v = 0.0f;
    if (j < H_SZ && k < K) v = W[(size_t)grow * K + k];
    Wc[(size_t)ncol * KP + k] = (_Float16)v;
}

__global__ void pack_bias(const float* __restrict__ bih,
                          const float* __restrict__ bhh,
                          float* __restrict__ bias) {
    int t = threadIdx.x;              // [0, 288) = 3 gates * 96
    if (t >= 3 * HP) return;
    int gate = t / HP, j = t % HP;
    int grow = (gate == 0) ? j : (gate == 1) ? (2 * H_SZ + j) : (3 * H_SZ + j);
    bias[t] = (j < H_SZ) ? (bih[grow] + bhh[grow]) : 0.0f;
}

// out_W [40, 21600] -> col-major fp16 [48][23040] with the 90->96 K padding
// matching h2's storage ([b][t*96+j]).
__global__ void pack_wout(const float* __restrict__ W, _Float16* __restrict__ Wc) {
    int n  = blockIdx.y;                                   // [0, 48)
    int kp = blockIdx.x * blockDim.x + threadIdx.x;        // [0, 23040)
    if (kp >= KOUT) return;
    int t = kp / HP, j = kp % HP;
    float v = 0.0f;
    if (n < 40 && j < H_SZ) v = W[(size_t)n * (T_SZ * H_SZ) + t * H_SZ + j];
    Wc[(size_t)n * KOUT + kp] = (_Float16)v;
}

// ---------------------------------------------------------------------------
// Fused GEMM + LSTM-cell(zero state) activation.
//   gates = A @ Wcol + bias ;  h = sig(o)*tanh(sig(i)*tanh(g)) [; sigmoid()]
// Block: 192 threads = 6 waves; wave w owns H-column block jb = w.
// A tile (16 rows x KP) staged in LDS; FROM_X transposes x on the fly,
// otherwise fp16 rows are DMA'd with GLOBAL_LOAD_ASYNC_TO_LDS_B128.
// ---------------------------------------------------------------------------
template <int KP, bool FROM_X, bool OUTER_SIG>
__global__ __launch_bounds__(192) void gemm_lstm(
    const void* __restrict__ Asrc, const _Float16* __restrict__ Wcol,
    const float* __restrict__ bias, _Float16* __restrict__ Hout) {

    __shared__ __align__(16) _Float16 ldsA[16 * KP];
    const int tid = threadIdx.x;
    const int rt  = blockIdx.x;       // row tile, 16 rows of M
    const int m0  = rt * 16;

    if (FROM_X) {
        // x layout [b][c][f][t]; A[m][k] = x[(b*180 + k)*240 + t], m=b*240+t.
        // Tile never crosses b (240 = 15*16).
        const float* x = (const float*)Asrc;
        const int b  = rt / 15;
        const int t0 = (rt % 15) * 16;
        for (int idx = tid; idx < 16 * KP; idx += 192) {
            int k  = idx >> 4;        // feature index
            int tt = idx & 15;        // row within tile (coalesced along t)
            float v = 0.0f;
            if (k < F_SZ) v = x[((size_t)b * F_SZ + k) * T_SZ + t0 + tt];
            ldsA[tt * KP + k] = (_Float16)v;
        }
    } else {
        const _Float16* hsrc = (const _Float16*)Asrc + (size_t)m0 * KP;
        const int chunks = 16 * KP / 8;          // 16-byte chunks
#if HAVE_ASYNC_LDS
        for (int idx = tid; idx < chunks; idx += 192)
            async_copy_b128(hsrc + idx * 8, ldsA + idx * 8);
        wait_async_zero();   // drain this wave's ASYNCcnt before the barrier
#else
        for (int idx = tid; idx < chunks; idx += 192)
            ((v8h*)ldsA)[idx] = ((const v8h*)hsrc)[idx];
#endif
    }
    __syncthreads();

    const int wave = tid >> 5;        // jb in [0,6)
    const int lane = tid & 31;
    const int hs   = (lane < 16) ? 0 : 1;   // half-select per ISA frag layout
    const int rc   = lane & 15;             // row (A) / column (B) in tile

    v8f ci = {}, cg = {}, co = {};
    const _Float16* wbase = Wcol + (size_t)(wave * 48) * KP;

#pragma unroll
    for (int ks = 0; ks < KP / 32; ++ks) {
        const int k0 = ks * 32;
        Frag16 A, Bi, Bg, Bo;
        const _Float16* ap = ldsA + rc * KP + k0 + hs * 8;
        A.h[0] = *(const v8h*)(ap);
        A.h[1] = *(const v8h*)(ap + 16);
        const _Float16* bi = wbase + ((size_t)(0 * 16 + rc)) * KP + k0 + hs * 8;
        const _Float16* bg = wbase + ((size_t)(1 * 16 + rc)) * KP + k0 + hs * 8;
        const _Float16* bo = wbase + ((size_t)(2 * 16 + rc)) * KP + k0 + hs * 8;
        Bi.h[0] = *(const v8h*)bi; Bi.h[1] = *(const v8h*)(bi + 16);
        Bg.h[0] = *(const v8h*)bg; Bg.h[1] = *(const v8h*)(bg + 16);
        Bo.h[0] = *(const v8h*)bo; Bo.h[1] = *(const v8h*)(bo + 16);
        ci = __builtin_amdgcn_wmma_f32_16x16x32_f16(false, A.v, false, Bi.v,
                                                    (short)0, ci, false, false);
        cg = __builtin_amdgcn_wmma_f32_16x16x32_f16(false, A.v, false, Bg.v,
                                                    (short)0, cg, false, false);
        co = __builtin_amdgcn_wmma_f32_16x16x32_f16(false, A.v, false, Bo.v,
                                                    (short)0, co, false, false);
    }

    const int ncol = wave * 16 + rc;        // H column in [0,96)
    const float bi = bias[0 * HP + ncol];
    const float bg = bias[1 * HP + ncol];
    const float bo = bias[2 * HP + ncol];
#pragma unroll
    for (int e = 0; e < 8; ++e) {
        float gi = ci[e] + bi, gg = cg[e] + bg, go = co[e] + bo;
        float c  = sigf(gi) * tanhf(gg);
        float h  = sigf(go) * tanhf(c);
        if (OUTER_SIG) h = sigf(h);
        int ml = e + hs * 8;                // C/D layout: M = vgpr + 8*(lane>=16)
        Hout[(size_t)(m0 + ml) * HP + ncol] = (_Float16)h;
    }
}

// ---------------------------------------------------------------------------
// Output projection, split-K WMMA. h2 viewed as [1024][23040] fp16 (A),
// Wout col-major [48][23040] (B). Partials -> P[SPLITS][1024][48] f32.
// One wave per block (EXEC all ones for WMMA).
// ---------------------------------------------------------------------------
__global__ __launch_bounds__(32) void out_gemm_partial(
    const _Float16* __restrict__ h2, const _Float16* __restrict__ Wout,
    float* __restrict__ P) {
    int lin   = blockIdx.x;          // 64 * 3 * SPLITS
    int split = lin % SPLITS; lin /= SPLITS;
    int nt    = lin % 3;      lin /= 3;
    int mt    = lin;                 // [0,64)
    const int lane = threadIdx.x;
    const int hs = lane >> 4, rc = lane & 15;

    const _Float16* arow = h2   + (size_t)(mt * 16 + rc) * KOUT;
    const _Float16* bcol = Wout + (size_t)(nt * 16 + rc) * KOUT;

    v8f acc = {};
    const int kbeg = split * 36 * 32;
    const int kend = kbeg + 36 * 32;
    for (int k0 = kbeg; k0 < kend; k0 += 32) {
        const _Float16* ap = arow + k0 + hs * 8;
        const _Float16* bp = bcol + k0 + hs * 8;
        __builtin_prefetch(ap + 256, 0, 0);   // global_prefetch_b8, stream K
        __builtin_prefetch(bp + 256, 0, 0);
        Frag16 A, Bf;
        A.h[0]  = *(const v8h*)ap; A.h[1]  = *(const v8h*)(ap + 16);
        Bf.h[0] = *(const v8h*)bp; Bf.h[1] = *(const v8h*)(bp + 16);
        acc = __builtin_amdgcn_wmma_f32_16x16x32_f16(false, A.v, false, Bf.v,
                                                     (short)0, acc, false, false);
    }
#pragma unroll
    for (int e = 0; e < 8; ++e) {
        int m = mt * 16 + e + hs * 8;
        int n = nt * 16 + rc;
        P[((size_t)split * B_SZ + m) * NOUTP + n] = acc[e];
    }
}

// ---------------------------------------------------------------------------
// Reduce split-K partials + bias, then softmax over groups of 10 (B,4,10).
// ---------------------------------------------------------------------------
__global__ __launch_bounds__(64) void reduce_softmax(
    const float* __restrict__ P, const float* __restrict__ out_b,
    float* __restrict__ out) {
    __shared__ float vals[40];
    __shared__ float gmax[4], gsum[4];
    const int b = blockIdx.x;
    const int n = threadIdx.x;
    if (n < 40) {
        float s = out_b[n];
        for (int sp = 0; sp < SPLITS; ++sp)
            s += P[((size_t)sp * B_SZ + b) * NOUTP + n];
        vals[n] = s;
    }
    __syncthreads();
    if (n < 4) {
        float m = vals[n * 10];
        for (int i = 1; i < 10; ++i) m = fmaxf(m, vals[n * 10 + i]);
        gmax[n] = m;
    }
    __syncthreads();
    if (n < 40) vals[n] = __expf(vals[n] - gmax[n / 10]);
    __syncthreads();
    if (n < 4) {
        float s = 0.0f;
        for (int i = 0; i < 10; ++i) s += vals[n * 10 + i];
        gsum[n] = s;
    }
    __syncthreads();
    if (n < 40) out[(size_t)b * 40 + n] = vals[n] / gsum[n / 10];
}

// ---------------------------------------------------------------------------
// Launch
// ---------------------------------------------------------------------------
extern "C" void kernel_launch(void* const* d_in, const int* in_sizes, int n_in,
                              void* d_out, int out_size, void* d_ws, size_t ws_size,
                              hipStream_t stream) {
    (void)in_sizes; (void)n_in; (void)out_size; (void)ws_size;
    const float* x       = (const float*)d_in[0];
    const float* enc_W   = (const float*)d_in[1];
    const float* enc_bih = (const float*)d_in[2];
    const float* enc_bhh = (const float*)d_in[3];
    const float* dec_W   = (const float*)d_in[4];
    const float* dec_bih = (const float*)d_in[5];
    const float* dec_bhh = (const float*)d_in[6];
    const float* out_W   = (const float*)d_in[7];
    const float* out_b   = (const float*)d_in[8];
    float* out = (float*)d_out;

    char* ws = (char*)d_ws;
    size_t off = 0;
    auto alloc = [&](size_t bytes) -> void* {
        off = (off + 255) & ~(size_t)255;
        void* p = ws + off;
        off += bytes;
        return p;
    };
    _Float16* Wenc  = (_Float16*)alloc((size_t)NCOL * KP_ENC * sizeof(_Float16));
    _Float16* Wdec  = (_Float16*)alloc((size_t)NCOL * KP_DEC * sizeof(_Float16));
    _Float16* Wout  = (_Float16*)alloc((size_t)NOUTP * KOUT * sizeof(_Float16));
    float*    biasE = (float*)alloc(3 * HP * sizeof(float));
    float*    biasD = (float*)alloc(3 * HP * sizeof(float));
    _Float16* h1    = (_Float16*)alloc((size_t)MROWS * HP * sizeof(_Float16));
    _Float16* h2    = (_Float16*)alloc((size_t)MROWS * HP * sizeof(_Float16));
    float*    P     = (float*)h1;   // h1 dead after decoder; reuse (3.9MB < 47MB)

    // 1) pack weights / biases
    pack_w<<<NCOL, KP_ENC, 0, stream>>>(enc_W, Wenc, F_SZ, KP_ENC);
    pack_w<<<NCOL, KP_DEC, 0, stream>>>(dec_W, Wdec, H_SZ, KP_DEC);
    pack_wout<<<dim3(KOUT / 256, NOUTP), 256, 0, stream>>>(out_W, Wout);
    pack_bias<<<1, 3 * HP, 0, stream>>>(enc_bih, enc_bhh, biasE);
    pack_bias<<<1, 3 * HP, 0, stream>>>(dec_bih, dec_bhh, biasD);

    // 2) encoder: x -> h1 (extra outer sigmoid)
    gemm_lstm<KP_ENC, true, true><<<MROWS / 16, 192, 0, stream>>>(
        (const void*)x, Wenc, biasE, h1);

    // 3) decoder: h1 -> h2 (A tile staged via async global->LDS DMA)
    gemm_lstm<KP_DEC, false, false><<<MROWS / 16, 192, 0, stream>>>(
        (const void*)h1, Wdec, biasD, h2);

    // 4) output projection (split-K) + softmax
    out_gemm_partial<<<64 * 3 * SPLITS, 32, 0, stream>>>(h2, Wout, P);
    reduce_softmax<<<B_SZ, 64, 0, stream>>>(P, out_b, out);
}
